// NeRF_19095424598660
// MI455X (gfx1250) — matrix operand order repaired and checked by reading, not compiled, not run on hardware
//
#include <hip/hip_runtime.h>
#include <hip/hip_bf16.h>

typedef __attribute__((ext_vector_type(16))) _Float16 v16h;
typedef __attribute__((ext_vector_type(8)))  float    v8f;
typedef __attribute__((ext_vector_type(4)))  unsigned int u32x4;
typedef __attribute__((ext_vector_type(8)))  int i32x8;
typedef __attribute__((ext_vector_type(4)))  int i32x4;

#define ROWS 128      // sample rows per block (16 points * 8 samples)
#define PTS  16
#define NS   8
#define FC   19
#define CH   23       // FEAT_CH + 4
#define RS   23       // raw stride (contiguous: TDM writes the tile linearly)

// ---- LDS byte offsets (manually carved from dynamic shared) ----
#define O_RAW    0        // f32[128*23]                 11776 (region 12288)
#define O_AFRAG  12288    // half[8 Mt][2 ks][32][16]    16384  (phase: stages 3-4)
#define O_XFRAG  12288    // half[8 Mt][4 ks][32][16]    32768  (phase: stages 7-8, overlays AFRAG+IFR)
#define O_IFR    28672    // f32[128*20]                 10240  (phase: stages 1-3)
#define O_G      45056    // f32[128*32]                 16384
#define O_GWF    61440    // half[2 ks][2 nt][32][16]     4096
#define O_C0WF   65536    // half[4 ks][4 nt][32][16]    16384
#define O_MEAN   81920    // f32[16*19]
#define O_VAR    83136    // f32[16*19]
#define O_AW     84352    // f32[128]
#define O_SMW    84864    // f32[128]
#define O_WSUM   85376    // f32[16*32]
#define O_IMF    87424    // f32[16*16]
#define O_X      88448    // f32[16*64]
#define O_SIG    92544    // f32[16]
#define O_CW     92608    // f32[128]
#define O_VW     93120    // f32[76]
#define O_VB     93424    // f32[19]
#define O_GB     93500    // f32[32]
#define O_AGW    93628    // f32[32]
#define O_AGWB   93756    // f32[1]
#define O_AFCW   93760    // f32[512]
#define O_AFCB   95808    // f32[16]
#define O_LR0W   95872    // f32[1024]
#define O_LR0B   99968    // f32[64]
#define O_SGW    100224   // f32[64]
#define O_SGB    100480   // f32[1]
#define O_C0B    100484   // f32[64]
#define O_C1W    100740   // f32[64]
#define O_C1B    100996   // f32[1]
#define SMEM_BYTES 101376

// A-fragment (16-bit, 16x32 per K-step) swizzle, contiguous K order (kk = 16*half + e, same as B):
// lane = (row%16) + 16*half ; VGPR/elem packing of K within a 32-wide step.
__device__ __forceinline__ int afrag_idx(int row, int k, int nks) {
  int mt = row >> 4, m = row & 15;
  int ks = k >> 5, kk = k & 31;
  int half = kk >> 4, e = kk & 15;
  int lane = m + (half << 4);
  return ((mt * nks + ks) * 32 + lane) * 16 + e;
}

__device__ __forceinline__ v8f wmma_f16(v16h a, v16h b, v8f c) {
  return __builtin_amdgcn_wmma_f32_16x16x32_f16(false, a, false, b, (short)0, c, false, false);
}

__global__ void __launch_bounds__(128, 1) nerf_fused(
    const float* __restrict__ img,
    const float* __restrict__ vw,   const float* __restrict__ vb,
    const float* __restrict__ gw,   const float* __restrict__ gb,
    const float* __restrict__ awW,  const float* __restrict__ awB,
    const float* __restrict__ afcW, const float* __restrict__ afcB,
    const float* __restrict__ lr0W, const float* __restrict__ lr0B,
    const float* __restrict__ sgW,  const float* __restrict__ sgB,
    const float* __restrict__ c0W,  const float* __restrict__ c0B,
    const float* __restrict__ c1W,  const float* __restrict__ c1B,
    float* __restrict__ out)
{
  extern __shared__ char smem[];
  const int tid  = threadIdx.x;
  const int lane = tid & 31;
  const int wv   = tid >> 5;

  float*     raw   = (float*)(smem + O_RAW);
  float*     ifr   = (float*)(smem + O_IFR);
  float*     gS    = (float*)(smem + O_G);
  _Float16*  afrag = (_Float16*)(smem + O_AFRAG);
  _Float16*  xfrag = (_Float16*)(smem + O_XFRAG);
  _Float16*  gwf   = (_Float16*)(smem + O_GWF);
  _Float16*  c0wf  = (_Float16*)(smem + O_C0WF);
  float*     meanS = (float*)(smem + O_MEAN);
  float*     varS  = (float*)(smem + O_VAR);
  float*     cwS   = (float*)(smem + O_CW);

  // ---- Stage 0a: wave 0 kicks off the TDM bulk copy of the activation tile
  // (2944 dwords, global -> LDS offset 0), overlapped with weight staging.
  if (wv == 0) {
    unsigned long long gaddr =
        (unsigned long long)(img + (long)blockIdx.x * (ROWS * CH));
    // D# group 0 (ISA 8.3): count=1 | lds_addr | global_addr[56:0] | type=2
    u32x4 g0;
    g0.x = 1u;                                   // count=1, user mode
    g0.y = (unsigned)O_RAW;                      // LDS byte address
    g0.z = (unsigned)(gaddr & 0xFFFFFFFFull);    // global_addr[31:0]
    g0.w = (unsigned)((gaddr >> 32) & 0x1FFFFFFull) | (2u << 30); // [56:32]|type=2
    // D# group 1 (ISA 8.4): 1-D copy, data_size=4B (code 2)
    i32x8 g1;
    g1[0] = (2 << 16);             // workgroup_mask=0 (not in cluster), data_size=2
    g1[1] = (2944 & 0xFFFF) << 16; // tensor_dim0[15:0] in bits 63:48
    g1[2] = (2944 >> 16) | (1 << 16); // tensor_dim0[31:16] ; tensor_dim1=1
    g1[3] = (2944 & 0xFFFF) << 16; // tile_dim0 = 2944 (bits 127:112)
    g1[4] = 0;                     // tile_dim1=0 (unused), tile_dim2=0
    g1[5] = 2944;                  // tensor_dim0_stride[31:0]
    g1[6] = 0;                     // stride0 hi, stride1 lo
    g1[7] = 0;
    i32x4 gz4 = {0, 0, 0, 0};
    i32x8 gz8 = {0, 0, 0, 0, 0, 0, 0, 0};
    __builtin_amdgcn_tensor_load_to_lds(g0, g1, gz4, gz4, gz8, 0);
  }

  // ---- Stage 0b: weight staging / pre-swizzle (all waves, overlaps TDM) ----
  {
#define CPY(off, src, n) { float* d_ = (float*)(smem + (off)); \
    for (int i_ = tid; i_ < (n); i_ += 128) d_[i_] = (src)[i_]; }
    CPY(O_VW,  vw,  76)  CPY(O_VB,  vb,  19)  CPY(O_GB,  gb,  32)
    CPY(O_AGW, awW, 32)  CPY(O_AGWB,awB, 1)
    CPY(O_AFCW,afcW,512) CPY(O_AFCB,afcB,16)
    CPY(O_LR0W,lr0W,1024)CPY(O_LR0B,lr0B,64)
    CPY(O_SGW, sgW, 64)  CPY(O_SGB, sgB, 1)
    CPY(O_C0B, c0B, 64)  CPY(O_C1W, c1W, 64)  CPY(O_C1B, c1B, 1)
#undef CPY

    // global_w (57x32, K-pad to 64) into WMMA B-fragment layout [ks][nt][lane][e]
    for (int i = tid; i < 2 * 2 * 32 * 16; i += 128) {
      int e = i & 15, L = (i >> 4) & 31, nt = (i >> 9) & 1, ks = i >> 10;
      int K = ks * 32 + ((L >> 4) << 4) + e;
      int n = nt * 16 + (L & 15);
      gwf[i] = (_Float16)((K < 57) ? gw[K * 32 + n] : 0.0f);
    }
    // color0_w (103x64, K-pad to 128) into B-fragment layout
    for (int i = tid; i < 4 * 4 * 32 * 16; i += 128) {
      int e = i & 15, L = (i >> 4) & 31, nt = (i >> 9) & 3, ks = i >> 11;
      int K = ks * 32 + ((L >> 4) << 4) + e;
      int n = nt * 16 + (L & 15);
      c0wf[i] = (_Float16)((K < 103) ? c0W[K * 64 + n] : 0.0f);
    }
    cwS[tid] = 0.0f;
  }
  // Issuing wave drains TENSORcnt, then the barrier publishes the tile.
  if (wv == 0) __builtin_amdgcn_s_wait_tensorcnt(0);
  __syncthreads();

  // ---- Stage 1: view MLP per row: ifr = feat + relu(dirs @ view_w + b) ----
  {
    const float* r  = raw + tid * RS;
    const float* VW = (const float*)(smem + O_VW);
    const float* VB = (const float*)(smem + O_VB);
    float d0 = r[19], d1 = r[20], d2 = r[21], d3 = r[22];
    for (int c = 0; c < FC; ++c) {
      float t = d0 * VW[c] + d1 * VW[19 + c] + d2 * VW[38 + c] + d3 * VW[57 + c] + VB[c];
      ifr[tid * 20 + c] = r[c] + (t > 0.f ? t : 0.f);
    }
  }
  __syncthreads();

  // ---- Stage 2: per-point mean / unbiased var over S ----
  for (int i = tid; i < PTS * FC; i += 128) {
    int p = i / FC, c = i % FC;
    float s = 0.f;
    #pragma unroll
    for (int k = 0; k < NS; ++k) s += ifr[(p * NS + k) * 20 + c];
    float mn = s * 0.125f, v = 0.f;
    #pragma unroll
    for (int k = 0; k < NS; ++k) { float d = ifr[(p * NS + k) * 20 + c] - mn; v += d * d; }
    meanS[i] = mn;
    varS[i]  = v * (1.0f / 7.0f);
  }
  __syncthreads();

  // ---- Stage 3: cat = [ifr | var | mean | 0] -> f16 A-fragment layout (K=64) ----
  for (int i = tid; i < ROWS * 64; i += 128) {
    int row = i >> 6, k = i & 63, p = row >> 3;
    float v;
    if      (k < 19) v = ifr[row * 20 + k];
    else if (k < 38) v = varS[p * FC + (k - 19)];
    else if (k < 57) v = meanS[p * FC + (k - 38)];
    else             v = 0.f;
    afrag[afrag_idx(row, k, 2)] = (_Float16)v;
  }
  __syncthreads();

  // ---- Stage 4: g = relu(cat @ global_w + gb) via WMMA (M=128,N=32,K=64) ----
  {
    const float* GB = (const float*)(smem + O_GB);
    for (int t = wv; t < 16; t += 4) {
      int mt = t >> 1, nt = t & 1;
      v8f acc = {};
      #pragma unroll
      for (int ks = 0; ks < 2; ++ks) {
        v16h a = *(const v16h*)(afrag + ((mt * 2 + ks) * 32 + lane) * 16);
        v16h b = *(const v16h*)(gwf   + ((ks * 2 + nt) * 32 + lane) * 16);
        acc = wmma_f16(a, b, acc);
      }
      int half = lane >> 4, col = nt * 16 + (lane & 15);
      float bias = GB[col];
      #pragma unroll
      for (int r = 0; r < 8; ++r) {
        int row = mt * 16 + half * 8 + r;
        float v = acc[r] + bias;
        gS[row * 32 + col] = v > 0.f ? v : 0.f;
      }
    }
  }
  __syncthreads();

  // ---- Stage 5: attention aggregate + aggfc ----
  {
    const float* AGW = (const float*)(smem + O_AGW);
    float awb = ((const float*)(smem + O_AGWB))[0];
    float* awS = (float*)(smem + O_AW);
    float acc = awb;
    for (int c = 0; c < 32; ++c) acc += gS[tid * 32 + c] * AGW[c];
    awS[tid] = acc > 0.f ? acc : 0.f;
  }
  __syncthreads();
  if (tid < PTS) {
    float* awS  = (float*)(smem + O_AW);
    float* smwS = (float*)(smem + O_SMW);
    float m = -1e30f;
    #pragma unroll
    for (int s = 0; s < NS; ++s) { float z = awS[tid * NS + s]; if (z > m) m = z; }
    float den = 0.f, e[NS];
    #pragma unroll
    for (int s = 0; s < NS; ++s) { e[s] = __expf(awS[tid * NS + s] - m); den += e[s]; }
    float inv = 1.0f / den;
    #pragma unroll
    for (int s = 0; s < NS; ++s) smwS[tid * NS + s] = e[s] * inv;
  }
  __syncthreads();
  {
    float* smwS = (float*)(smem + O_SMW);
    float* wsum = (float*)(smem + O_WSUM);
    for (int i = tid; i < PTS * 32; i += 128) {
      int p = i >> 5, c = i & 31;
      float s = 0.f;
      #pragma unroll
      for (int k = 0; k < NS; ++k) s += gS[(p * NS + k) * 32 + c] * smwS[p * NS + k];
      wsum[i] = s;
    }
  }
  __syncthreads();
  {
    const float* AFC = (const float*)(smem + O_AFCW);
    const float* AFB = (const float*)(smem + O_AFCB);
    float* wsum = (float*)(smem + O_WSUM);
    float* imf  = (float*)(smem + O_IMF);
    for (int i = tid; i < PTS * 16; i += 128) {
      int p = i >> 4, o = i & 15;
      float s = AFB[o];
      for (int c = 0; c < 32; ++c) s += wsum[p * 32 + c] * AFC[c * 16 + o];
      imf[i] = s > 0.f ? s : 0.f;
    }
  }
  __syncthreads();

  // ---- Stage 6: x = relu(imf @ lr0) ; sigma = softplus(x @ sigma_w) ----
  {
    const float* L0 = (const float*)(smem + O_LR0W);
    const float* LB = (const float*)(smem + O_LR0B);
    float* imf = (float*)(smem + O_IMF);
    float* xS  = (float*)(smem + O_X);
    for (int i = tid; i < PTS * 64; i += 128) {
      int p = i >> 6, o = i & 63;
      float s = LB[o];
      #pragma unroll
      for (int c = 0; c < 16; ++c) s += imf[p * 16 + c] * L0[c * 64 + o];
      xS[i] = s > 0.f ? s : 0.f;
    }
  }
  __syncthreads();
  if (tid < PTS) {
    const float* SW = (const float*)(smem + O_SGW);
    float* xS = (float*)(smem + O_X);
    float z = ((const float*)(smem + O_SGB))[0];
    for (int o = 0; o < 64; ++o) z += xS[tid * 64 + o] * SW[o];
    ((float*)(smem + O_SIG))[tid] = (z > 20.f) ? z : log1pf(__expf(z));
  }
  // ---- Stage 7: xc = [x | imf | raw23 | 0] -> f16 A-fragment layout (K=128) ----
  {
    float* imf = (float*)(smem + O_IMF);
    float* xS  = (float*)(smem + O_X);
    for (int i = tid; i < ROWS * 128; i += 128) {
      int row = i >> 7, k = i & 127, p = row >> 3;
      float v;
      if      (k < 64)  v = xS[p * 64 + k];
      else if (k < 80)  v = imf[p * 16 + (k - 64)];
      else if (k < 103) v = raw[row * RS + (k - 80)];
      else              v = 0.f;
      xfrag[afrag_idx(row, k, 4)] = (_Float16)v;
    }
  }
  __syncthreads();

  // ---- Stage 8: h=relu(xc@color0+b); fuse color1 dot via LDS atomics ----
  {
    const float* C0B = (const float*)(smem + O_C0B);
    const float* C1W = (const float*)(smem + O_C1W);
    for (int t = wv; t < 32; t += 4) {
      int mt = t >> 2, nt = t & 3;
      v8f acc = {};
      #pragma unroll
      for (int ks = 0; ks < 4; ++ks) {
        v16h a = *(const v16h*)(xfrag + ((mt * 4 + ks) * 32 + lane) * 16);
        v16h b = *(const v16h*)(c0wf  + ((ks * 4 + nt) * 32 + lane) * 16);
        acc = wmma_f16(a, b, acc);
      }
      int half = lane >> 4, col = nt * 16 + (lane & 15);
      float bias = C0B[col], w1 = C1W[col];
      #pragma unroll
      for (int r = 0; r < 8; ++r) {
        int row = mt * 16 + half * 8 + r;
        float h = acc[r] + bias;
        h = h > 0.f ? h : 0.f;
        atomicAdd(&cwS[row], h * w1);
      }
    }
  }
  __syncthreads();

  // ---- Stage 9: per-point softmax over S, RGB blend, emit [color|sigma] ----
  if (tid < PTS) {
    float c1b = ((const float*)(smem + O_C1B))[0];
    float v[NS], m = -1e30f;
    #pragma unroll
    for (int s = 0; s < NS; ++s) {
      float z = cwS[tid * NS + s] + c1b;
      z = z > 0.f ? z : 0.f;
      v[s] = z; if (z > m) m = z;
    }
    float den = 0.f;
    #pragma unroll
    for (int s = 0; s < NS; ++s) { v[s] = __expf(v[s] - m); den += v[s]; }
    float inv = 1.0f / den, r0 = 0.f, r1 = 0.f, r2 = 0.f;
    #pragma unroll
    for (int s = 0; s < NS; ++s) {
      float wgt = v[s] * inv;
      const float* r = raw + (tid * NS + s) * RS;
      r0 += r[16] * wgt; r1 += r[17] * wgt; r2 += r[18] * wgt;
    }
    long gp = (long)blockIdx.x * PTS + tid;
    out[gp * 4 + 0] = r0;
    out[gp * 4 + 1] = r1;
    out[gp * 4 + 2] = r2;
    out[gp * 4 + 3] = ((const float*)(smem + O_SIG))[tid];
  }
}

extern "C" void kernel_launch(void* const* d_in, const int* in_sizes, int n_in,
                              void* d_out, int out_size, void* d_ws, size_t ws_size,
                              hipStream_t stream) {
  (void)d_ws; (void)ws_size; (void)n_in; (void)out_size;
  const float* img  = (const float*)d_in[1];
  const float* vw   = (const float*)d_in[2];
  const float* vb   = (const float*)d_in[3];
  const float* gw   = (const float*)d_in[4];
  const float* gb   = (const float*)d_in[5];
  const float* awW  = (const float*)d_in[6];
  const float* awB  = (const float*)d_in[7];
  const float* afcW = (const float*)d_in[8];
  const float* afcB = (const float*)d_in[9];
  const float* lr0W = (const float*)d_in[10];
  const float* lr0B = (const float*)d_in[11];
  const float* sgW  = (const float*)d_in[12];
  const float* sgB  = (const float*)d_in[13];
  const float* c0W  = (const float*)d_in[14];
  const float* c0B  = (const float*)d_in[15];
  const float* c1W  = (const float*)d_in[16];
  const float* c1B  = (const float*)d_in[17];
  float* out = (float*)d_out;

  int npts   = in_sizes[1] / (NS * CH);   // B*P = 65536
  int blocks = npts / PTS;                // 4096

  (void)hipFuncSetAttribute((const void*)nerf_fused,
                            hipFuncAttributeMaxDynamicSharedMemorySize, SMEM_BYTES);
  nerf_fused<<<blocks, 128, SMEM_BYTES, stream>>>(
      img, vw, vb, gw, gb, awW, awB, afcW, afcB,
      lr0W, lr0B, sgW, sgB, c0W, c0B, c1W, c1B, out);
}
